// Transformer_16655883174042
// MI455X (gfx1250) — compile-verified, matrix operand order checked by reading, not run on hardware
//
#include <hip/hip_runtime.h>
#include <cstdint>
#include <cstddef>

typedef _Float16 h16;
typedef __attribute__((ext_vector_type(16))) _Float16 v16h;
typedef __attribute__((ext_vector_type(8)))  _Float16 v8h;
typedef __attribute__((ext_vector_type(4)))  _Float16 v4h;
typedef __attribute__((ext_vector_type(8)))  float    v8f;
typedef __attribute__((ext_vector_type(4)))  float    v4f;

#define B_  2
#define S_  2048
#define D_  768
#define H_  12
#define DH_ 64
#define DF_ 3072
#define V_  32000
#define L_  2
#define BS_ (B_*S_)

// ---- WMMA helpers (CDNA5 gfx1250, wave32) -------------------------------
// A/B fragment layout per ISA 7.12.2: lane lo = row(M)/col(N), halves hold
// K = 8*hi .. 8*hi+7 and K = 8*hi+16 .. 8*hi+23.
__device__ __forceinline__ v16h frag_ld(const h16* p) {
    v8h a = *(const v8h*)p;
    v8h b = *(const v8h*)(p + 16);
    v16h r;
#pragma unroll
    for (int i = 0; i < 8; ++i) { r[i] = a[i]; r[i + 8] = b[i]; }
    return r;
}

__device__ __forceinline__ v8f wmma16(v16h a, v16h b, v8f c) {
    return __builtin_amdgcn_wmma_f32_16x16x32_f16(false, a, false, b,
                                                  (short)0, c, false, false);
}

__device__ __forceinline__ float gelu_exact(float x) {
    return 0.5f * x * (1.0f + erff(x * 0.7071067811865475f));
}

// ---- CDNA5 async global->LDS copy (ASYNCcnt-tracked) ---------------------
// GVS mode: mem_addr = SGPR64 + VGPR_i32 offset; VDST VGPR = LDS byte addr.
__device__ __forceinline__ void async_b128(uint32_t lds_off, uint32_t voff,
                                           uint64_t sbase) {
    asm volatile("global_load_async_to_lds_b128 %0, %1, %2"
                 :: "v"(lds_off), "v"(voff), "s"(sbase) : "memory");
}
__device__ __forceinline__ void wait_async0() {
    asm volatile("s_wait_asynccnt 0x0" ::: "memory");
}

// ---- f32 -> f16 convert (x4 vectorized; all sizes are multiples of 4) ----
__global__ void cvt_f32_f16(const float* __restrict__ in, h16* __restrict__ out, int n4) {
    int i = blockIdx.x * 256 + threadIdx.x;
    if (i < n4) {
        v4f a = ((const v4f*)in)[i];
        v4h b;
#pragma unroll
        for (int j = 0; j < 4; ++j) b[j] = (h16)a[j];
        ((v4h*)out)[i] = b;
    }
}

// ---- f32 [K][N] -> f16 [N][K] transpose-convert (32x32 tiles) ------------
__global__ __launch_bounds__(256) void cvt_transpose(const float* __restrict__ W,
                                                     h16* __restrict__ Wt,
                                                     int K, int N) {
    __shared__ float t[32][33];
    int k0 = blockIdx.y * 32, n0 = blockIdx.x * 32;
    int tx = threadIdx.x & 31, ty = threadIdx.x >> 5;   // 32 x 8
#pragma unroll
    for (int i = 0; i < 32; i += 8)
        t[ty + i][tx] = W[(size_t)(k0 + ty + i) * N + n0 + tx];
    __syncthreads();
#pragma unroll
    for (int i = 0; i < 32; i += 8)
        Wt[(size_t)(n0 + ty + i) * K + k0 + tx] = (h16)t[tx][ty + i];
}

// ---- embedding gather + sinusoidal positional encoding -------------------
__global__ void embed_kernel(const int* __restrict__ ids, const float* __restrict__ E,
                             float* __restrict__ x) {
    int t = blockIdx.x * 256 + threadIdx.x;
    if (t >= BS_ * D_) return;
    int d  = t % D_;
    int bs = t / D_;
    int s  = bs % S_;
    int id = ids[bs];
    int ip = d >> 1;
    float e   = (4.0f * (float)ip) / (float)D_;        // 2*i/D with i = 2*ip
    float ang = (float)s * __powf(10000.0f, -e);
    float pe  = (d & 1) ? __cosf(ang) : __sinf(ang);
    x[t] = E[(size_t)id * D_ + d] + pe;
}

// ---- tiled WMMA GEMM: C[M,N] = act(A16[M,K] @ Bt16[N,K]^T + bias) --------
// Both A and B tiles are straight 16B async global->LDS copies, double
// buffered; one barrier + s_wait_asynccnt per 32-wide K step.
template <int BIASF, int GELUF, int OUTH>
__global__ __launch_bounds__(256) void gemm_f16k(const h16* __restrict__ A,
                                                 const h16* __restrict__ Bt,
                                                 const float* __restrict__ bias,
                                                 void* __restrict__ Cout,
                                                 int M, int N, int K) {
    __shared__ h16 As[2][128 * 48];   // [row][k], stride 48 halves (96B)
    __shared__ h16 Bs[2][128 * 48];   // [n][k]

    const int tid  = threadIdx.x;
    const int lane = tid & 31, w = tid >> 5;
    const int lo   = lane & 15, hi = lane >> 4;
    const int wm   = w & 3,  wn = w >> 2;              // 4x2 waves -> 128x128
    const int m0   = blockIdx.y * 128, n0 = blockIdx.x * 128;

    // this thread's two 8-half chunks of the 128x32 tile
    const int c0   = tid * 2;
    const int row0 = c0 >> 2,       col0 = (c0 & 3) * 8;
    const int row1 = (c0 + 1) >> 2, col1 = ((c0 + 1) & 3) * 8;
    const uint32_t voff0 = (uint32_t)((row0 * K + col0) * 2);
    const uint32_t voff1 = (uint32_t)((row1 * K + col1) * 2);
    const uint64_t abase = (uint64_t)(uintptr_t)(A  + (size_t)m0 * K);
    const uint64_t bbase = (uint64_t)(uintptr_t)(Bt + (size_t)n0 * K);

    auto issue = [&](int nb, int kk) {
        uint64_t sA = abase + (uint64_t)kk * 2;
        uint64_t sB = bbase + (uint64_t)kk * 2;
        async_b128((uint32_t)(uintptr_t)&As[nb][row0 * 48 + col0], voff0, sA);
        async_b128((uint32_t)(uintptr_t)&As[nb][row1 * 48 + col1], voff1, sA);
        async_b128((uint32_t)(uintptr_t)&Bs[nb][row0 * 48 + col0], voff0, sB);
        async_b128((uint32_t)(uintptr_t)&Bs[nb][row1 * 48 + col1], voff1, sB);
    };

    v8f acc[2][4];
    const v8f vz = {0.f, 0.f, 0.f, 0.f, 0.f, 0.f, 0.f, 0.f};
#pragma unroll
    for (int i = 0; i < 2; ++i)
#pragma unroll
        for (int j = 0; j < 4; ++j) acc[i][j] = vz;

    issue(0, 0);
    const int nsteps = K >> 5;
    for (int s = 0; s < nsteps; ++s) {
        const int cur = s & 1;
        wait_async0();          // current buffer's copies landed (this wave)
        __syncthreads();        // ... and everyone else's; prev readers done
        if (s + 1 < nsteps) issue(cur ^ 1, (s + 1) * 32);  // overlap w/ WMMA

        v16h af[2], bf[4];
#pragma unroll
        for (int tm = 0; tm < 2; ++tm)
            af[tm] = frag_ld(&As[cur][(wm * 32 + tm * 16 + lo) * 48 + 8 * hi]);
#pragma unroll
        for (int tn = 0; tn < 4; ++tn)
            bf[tn] = frag_ld(&Bs[cur][(wn * 64 + tn * 16 + lo) * 48 + 8 * hi]);
#pragma unroll
        for (int tm = 0; tm < 2; ++tm)
#pragma unroll
            for (int tn = 0; tn < 4; ++tn)
                acc[tm][tn] = wmma16(af[tm], bf[tn], acc[tm][tn]);
    }

    // epilogue: C/D layout row = r + 8*hi, col = lo (per 16x16 tile)
#pragma unroll
    for (int tm = 0; tm < 2; ++tm) {
#pragma unroll
        for (int tn = 0; tn < 4; ++tn) {
            int col  = n0 + wn * 64 + tn * 16 + lo;
            float bv = BIASF ? bias[col] : 0.0f;
#pragma unroll
            for (int r = 0; r < 8; ++r) {
                int row  = m0 + wm * 32 + tm * 16 + r + 8 * hi;
                float vv = acc[tm][tn][r] + bv;
                if (GELUF) vv = gelu_exact(vv);
                if (OUTH) ((h16*)Cout)[(size_t)row * N + col] = (h16)vv;
                else      ((float*)Cout)[(size_t)row * N + col] = vv;
            }
        }
    }
}

// ---- flash attention: softmax(Q K^T / sqrt(D)) V -------------------------
// K tile is a straight async global->LDS copy (it is already B^T layout for
// QK^T); V needs a 16-bit transpose so it stays on the VGPR path.
__global__ __launch_bounds__(256) void attn_kernel(const h16* __restrict__ q,
                                                   const h16* __restrict__ k,
                                                   const h16* __restrict__ v,
                                                   h16* __restrict__ o) {
    __shared__ h16 Kt[64 * 80];        // [key][dh]   == B^T layout for QK^T
    __shared__ h16 Vt[64 * 80];        // [dh][key]   == B^T layout for PV
    __shared__ h16 Pw[8 * 16 * 80];    // per-wave P tile [row][key]

    const int tid = threadIdx.x, w = tid >> 5, lane = tid & 31;
    const int lo = lane & 15, hi = lane >> 4;
    const int bb = blockIdx.z, hh = blockIdx.y;
    const int qr = blockIdx.x * 128 + w * 16;
    const float scl = 0.03608439182435161f;            // 1/sqrt(768)

    v16h qf[2];
    {
        const h16* qp = q + (((size_t)bb * S_ + qr + lo) * H_ + hh) * DH_ + 8 * hi;
        qf[0] = frag_ld(qp);
        qf[1] = frag_ld(qp + 32);
    }

    const v8f vz = {0.f, 0.f, 0.f, 0.f, 0.f, 0.f, 0.f, 0.f};
    v8f oacc[4];
#pragma unroll
    for (int tn = 0; tn < 4; ++tn) oacc[tn] = vz;
    float m[8], l[8];
#pragma unroll
    for (int r = 0; r < 8; ++r) { m[r] = -__builtin_inff(); l[r] = 0.f; }

    h16* Pb = Pw + w * 16 * 80;

    // per-thread chunk coords for the 64x64 tiles (512 chunks of 8 halves)
    const int cA = tid * 2;
    const int key0 = cA >> 3,       dc0 = (cA & 7) * 8;
    const int key1 = (cA + 1) >> 3, dc1 = ((cA + 1) & 7) * 8;
    const uint32_t kvoff0 = (uint32_t)((key0 * H_ * DH_ + dc0) * 2);
    const uint32_t kvoff1 = (uint32_t)((key1 * H_ * DH_ + dc1) * 2);

    for (int kb = 0; kb < S_ / 64; ++kb) {
        __syncthreads();
        int ks = kb * 64;
        const uint64_t kbase =
            (uint64_t)(uintptr_t)(k + (((size_t)bb * S_ + ks) * H_ + hh) * DH_);
        async_b128((uint32_t)(uintptr_t)&Kt[key0 * 80 + dc0], kvoff0, kbase);
        async_b128((uint32_t)(uintptr_t)&Kt[key1 * 80 + dc1], kvoff1, kbase);
#pragma unroll
        for (int i = 0; i < 2; ++i) {
            int c = tid * 2 + i;
            int key = c >> 3, dc = (c & 7) * 8;
            size_t gi = (((size_t)bb * S_ + ks + key) * H_ + hh) * DH_ + dc;
            v8h vv = *(const v8h*)(v + gi);
#pragma unroll
            for (int j = 0; j < 8; ++j) Vt[(dc + j) * 80 + key] = vv[j];
        }
        wait_async0();
        __syncthreads();

        v8f sc[4];
#pragma unroll
        for (int tn = 0; tn < 4; ++tn) {
            sc[tn] = vz;
            v16h b0 = frag_ld(Kt + (tn * 16 + lo) * 80 + 8 * hi);
            v16h b1 = frag_ld(Kt + (tn * 16 + lo) * 80 + 32 + 8 * hi);
            sc[tn] = wmma16(qf[0], b0, sc[tn]);
            sc[tn] = wmma16(qf[1], b1, sc[tn]);
        }
#pragma unroll
        for (int tn = 0; tn < 4; ++tn)
#pragma unroll
            for (int r = 0; r < 8; ++r) sc[tn][r] *= scl;

        float corr[8];
#pragma unroll
        for (int r = 0; r < 8; ++r) {
            float rm = fmaxf(fmaxf(sc[0][r], sc[1][r]), fmaxf(sc[2][r], sc[3][r]));
            for (int off = 1; off < 16; off <<= 1) rm = fmaxf(rm, __shfl_xor(rm, off));
            float mn = fmaxf(m[r], rm);
            corr[r] = __expf(m[r] - mn);
            m[r] = mn;
            float rs = 0.f;
#pragma unroll
            for (int tn = 0; tn < 4; ++tn) {
                float p = __expf(sc[tn][r] - mn);
                sc[tn][r] = p;
                rs += p;
            }
            for (int off = 1; off < 16; off <<= 1) rs += __shfl_xor(rs, off);
            l[r] = l[r] * corr[r] + rs;
        }
#pragma unroll
        for (int tn = 0; tn < 4; ++tn)
#pragma unroll
            for (int r = 0; r < 8; ++r) {
                oacc[tn][r] *= corr[r];
                Pb[(r + 8 * hi) * 80 + tn * 16 + lo] = (h16)sc[tn][r];
            }
        __syncthreads();

        v16h pa0 = frag_ld(Pb + lo * 80 + 8 * hi);
        v16h pa1 = frag_ld(Pb + lo * 80 + 32 + 8 * hi);
#pragma unroll
        for (int tn = 0; tn < 4; ++tn) {
            v16h bv0 = frag_ld(Vt + (tn * 16 + lo) * 80 + 8 * hi);
            v16h bv1 = frag_ld(Vt + (tn * 16 + lo) * 80 + 32 + 8 * hi);
            oacc[tn] = wmma16(pa0, bv0, oacc[tn]);
            oacc[tn] = wmma16(pa1, bv1, oacc[tn]);
        }
    }

#pragma unroll
    for (int tn = 0; tn < 4; ++tn)
#pragma unroll
        for (int r = 0; r < 8; ++r) {
            int row = qr + r + 8 * hi;
            float val = oacc[tn][r] / l[r];
            o[((size_t)bb * S_ + row) * D_ + hh * DH_ + tn * 16 + lo] = (h16)val;
        }
}

// ---- LayerNorm (+optional residual add), one block per row (D=768) -------
__global__ __launch_bounds__(256) void ln_kernel(const float* __restrict__ xin,
                                                 const float* __restrict__ add,
                                                 const float* __restrict__ g,
                                                 const float* __restrict__ b,
                                                 float* __restrict__ xout) {
    __shared__ float red[256];
    int row = blockIdx.x;
    int tid = threadIdx.x;
    const float* xr = xin + (size_t)row * D_;
    const float* ar = add ? add + (size_t)row * D_ : nullptr;
    float vv[3];
    float s = 0.f;
#pragma unroll
    for (int i = 0; i < 3; ++i) {
        int c = tid + i * 256;
        float t = xr[c] + (ar ? ar[c] : 0.f);
        vv[i] = t;
        s += t;
    }
    red[tid] = s;
    __syncthreads();
    for (int o = 128; o > 0; o >>= 1) { if (tid < o) red[tid] += red[tid + o]; __syncthreads(); }
    float mean = red[0] / (float)D_;
    __syncthreads();
    float s2 = 0.f;
#pragma unroll
    for (int i = 0; i < 3; ++i) { float d = vv[i] - mean; s2 += d * d; }
    red[tid] = s2;
    __syncthreads();
    for (int o = 128; o > 0; o >>= 1) { if (tid < o) red[tid] += red[tid + o]; __syncthreads(); }
    float inv = rsqrtf(red[0] / (float)D_ + 1e-5f);
    float* xo = xout + (size_t)row * D_;
#pragma unroll
    for (int i = 0; i < 3; ++i) {
        int c = tid + i * 256;
        xo[c] = (vv[i] - mean) * inv * g[c] + b[c];
    }
}

// ---- orchestration -------------------------------------------------------
extern "C" void kernel_launch(void* const* d_in, const int* in_sizes, int n_in,
                              void* d_out, int out_size, void* d_ws, size_t ws_size,
                              hipStream_t stream) {
    (void)in_sizes; (void)n_in; (void)out_size; (void)ws_size;

    const int*   ids = (const int*)d_in[0];
    const float* E   = (const float*)d_in[1];
    const float* Wq  = (const float*)d_in[2];  const float* bq  = (const float*)d_in[3];
    const float* Wk  = (const float*)d_in[4];  const float* bk  = (const float*)d_in[5];
    const float* Wv  = (const float*)d_in[6];  const float* bv  = (const float*)d_in[7];
    const float* Wo  = (const float*)d_in[8];  const float* bo  = (const float*)d_in[9];
    const float* g1  = (const float*)d_in[10]; const float* be1 = (const float*)d_in[11];
    const float* W1  = (const float*)d_in[12]; const float* b1  = (const float*)d_in[13];
    const float* W2  = (const float*)d_in[14]; const float* b2  = (const float*)d_in[15];
    const float* g2  = (const float*)d_in[16]; const float* be2 = (const float*)d_in[17];
    const float* Wd  = (const float*)d_in[18]; const float* bd  = (const float*)d_in[19];
    const float* gm  = (const float*)d_in[20]; const float* bm  = (const float*)d_in[21];
    const float* db  = (const float*)d_in[22];
    float* out = (float*)d_out;

    char* ws = (char*)d_ws;
    size_t off = 0;
    auto alloc = [&](size_t bytes) -> char* {
        char* p = ws + off;
        off = (off + bytes + 255) & ~(size_t)255;
        return p;
    };
    h16*   E16  = (h16*)alloc((size_t)V_ * D_ * 2);
    h16*   wbuf = (h16*)alloc((size_t)D_ * DF_ * 2);   // one transposed weight
    float* x    = (float*)alloc((size_t)BS_ * D_ * 4);
    h16*   x16  = (h16*)alloc((size_t)BS_ * D_ * 2);
    h16*   q16  = (h16*)alloc((size_t)BS_ * D_ * 2);
    h16*   k16  = (h16*)alloc((size_t)BS_ * D_ * 2);
    h16*   v16  = (h16*)alloc((size_t)BS_ * D_ * 2);
    h16*   o16  = (h16*)alloc((size_t)BS_ * D_ * 2);
    float* y    = (float*)alloc((size_t)BS_ * D_ * 4);
    h16*   hbuf = (h16*)alloc((size_t)BS_ * DF_ * 2);

    auto cvt = [&](const float* src, h16* dst, size_t n) {
        int n4 = (int)(n / 4);
        cvt_f32_f16<<<dim3((unsigned)((n4 + 255) / 256)), dim3(256), 0, stream>>>(src, dst, n4);
    };
    auto cvtT = [&](const float* W, h16* Wt, int K, int N) {   // [K][N] -> [N][K] f16
        cvt_transpose<<<dim3(N / 32, K / 32), dim3(256), 0, stream>>>(W, Wt, K, N);
    };

    dim3 blk(256);
    dim3 gD(D_ / 128,  BS_ / 128);
    dim3 gF(DF_ / 128, BS_ / 128);
    dim3 gV(V_ / 128,  BS_ / 128);

    embed_kernel<<<dim3((BS_ * D_ + 255) / 256), blk, 0, stream>>>(ids, E, x);
    cvt(E, E16, (size_t)V_ * D_);   // embedding already [N=V][K=D] for the head

    for (int l = 0; l < L_; ++l) {
        cvt(x, x16, (size_t)BS_ * D_);
        cvtT(Wq + (size_t)l * D_ * D_, wbuf, D_, D_);
        gemm_f16k<1, 0, 1><<<gD, blk, 0, stream>>>(x16, wbuf, bq + l * D_, q16, BS_, D_, D_);
        cvtT(Wk + (size_t)l * D_ * D_, wbuf, D_, D_);
        gemm_f16k<1, 0, 1><<<gD, blk, 0, stream>>>(x16, wbuf, bk + l * D_, k16, BS_, D_, D_);
        cvtT(Wv + (size_t)l * D_ * D_, wbuf, D_, D_);
        gemm_f16k<1, 0, 1><<<gD, blk, 0, stream>>>(x16, wbuf, bv + l * D_, v16, BS_, D_, D_);

        attn_kernel<<<dim3(S_ / 128, H_, B_), blk, 0, stream>>>(q16, k16, v16, o16);

        cvtT(Wo + (size_t)l * D_ * D_, wbuf, D_, D_);
        gemm_f16k<1, 0, 0><<<gD, blk, 0, stream>>>(o16, wbuf, bo + l * D_, y, BS_, D_, D_);
        ln_kernel<<<BS_, blk, 0, stream>>>(x, y, g1 + l * D_, be1 + l * D_, x);

        cvt(x, x16, (size_t)BS_ * D_);
        cvtT(W1 + (size_t)l * D_ * DF_, wbuf, D_, DF_);
        gemm_f16k<1, 1, 1><<<gF, blk, 0, stream>>>(x16, wbuf, b1 + l * DF_, hbuf, BS_, DF_, D_);
        cvtT(W2 + (size_t)l * DF_ * D_, wbuf, DF_, D_);
        gemm_f16k<1, 0, 0><<<gD, blk, 0, stream>>>(hbuf, wbuf, b2 + l * D_, y, BS_, D_, DF_);
        ln_kernel<<<BS_, blk, 0, stream>>>(x, y, g2 + l * D_, be2 + l * D_, x);
    }

    // MLM head: gelu(x @ Wd + bd) -> LN -> @ E^T + dec_b
    cvt(x, x16, (size_t)BS_ * D_);
    cvtT(Wd, wbuf, D_, D_);
    gemm_f16k<1, 1, 0><<<gD, blk, 0, stream>>>(x16, wbuf, bd, y, BS_, D_, D_);
    ln_kernel<<<BS_, blk, 0, stream>>>(y, nullptr, gm, bm, y);
    cvt(y, x16, (size_t)BS_ * D_);
    gemm_f16k<1, 0, 0><<<gV, blk, 0, stream>>>(x16, E16, db, out, BS_, V_, D_);
}
